// UnifiedAttention_80221399155059
// MI455X (gfx1250) — matrix operand order repaired and checked by reading, not compiled, not run on hardware
//
#include <hip/hip_runtime.h>
#include <hip/hip_bf16.h>
#include <math.h>

typedef __attribute__((ext_vector_type(16))) __bf16 v16bf;
typedef __attribute__((ext_vector_type(8)))  __bf16 v8bf;
typedef __attribute__((ext_vector_type(8)))  float  v8f;
typedef __attribute__((ext_vector_type(4)))  unsigned int v4u;
typedef __attribute__((ext_vector_type(8)))  int  v8i;
typedef __attribute__((ext_vector_type(4)))  int  v4i;

#if defined(__has_builtin)
#if __has_builtin(__builtin_amdgcn_tensor_load_to_lds)
#define HAVE_TDM 1
#endif
#endif
#ifndef HAVE_TDM
#define HAVE_TDM 0
#endif

#define DEVI static __device__ __forceinline__

DEVI v16bf cat8(v8bf lo, v8bf hi) {
  return __builtin_shufflevector(lo, hi, 0,1,2,3,4,5,6,7,8,9,10,11,12,13,14,15);
}
DEVI float wave_sum(float v) {
#pragma unroll
  for (int o = 16; o > 0; o >>= 1) v += __shfl_xor(v, o, 32);
  return v;
}

// ---------------------------------------------------------------------------
// Generic tiled GEMM:  C[M,N] = A[M,K] (bf16, row-major lda) x W[N,K]^T (bf16)
// Block = 256 threads = 8 waves; block tile 32(M) x 256(N); wave tile 16x64.
// Weight tile (256 rows x 32 K, 16KB) is staged to LDS by the Tensor Data
// Mover (double buffered, wave0 issues, TENSORcnt + barrier sync), with a
// 16B-per-row LDS pad (TDM pad feature) to spread LDS banks.
// EPI: 0 = f32 out (+bias),  1 = f32 out (+bias +resid),  2 = GEGLU -> bf16.
// ---------------------------------------------------------------------------
template <int EPI>
__global__ __launch_bounds__(256) void gemm_bf16_kernel(
    const __bf16* __restrict__ A, int lda,
    const __bf16* __restrict__ W, int ldw, int Ntot,
    const float* __restrict__ bias,
    const float* __restrict__ resid, int ldr,
    const float* __restrict__ pre, int ldp,
    float* __restrict__ Cf, __bf16* __restrict__ Cb, int ldc, int K) {
  __shared__ alignas(64) char wt[2][256 * 80];  // 2 x (256 rows x (64B data + 16B pad))
  const int lane = threadIdx.x & 31;
  const int w    = threadIdx.x >> 5;
  const int li   = lane & 15;
  const int hi   = lane >> 4;
  const int m0   = blockIdx.y * 32 + (w >> 2) * 16;
  const int nblk = blockIdx.x * 256;
  const int nw   = (w & 3) * 64;

  const __bf16* Arow = A + (size_t)(m0 + li) * lda;
  v8f acc[4] = {v8f{}, v8f{}, v8f{}, v8f{}};

#if HAVE_TDM
  // --- Tensor DMA descriptor, static group1 (2D tile of bf16) --------------
  v8i g1;
  g1[0] = (int)((1u << 16) |              // data_size = 2 bytes
                (1u << 20) |              // pad_enable
                (3u << 22) |              // pad_interval: 16 DWORDs (64B) of data
                (3u << 25));              // pad_amount:   4 DWORDs (16B) of pad
  g1[1] = (int)(((unsigned)K & 0xFFFFu) << 16);                       // tensor_dim0 lo
  g1[2] = (int)(((unsigned)K >> 16) | (((unsigned)Ntot & 0xFFFFu) << 16));
  g1[3] = (int)(((unsigned)Ntot >> 16) | (32u << 16));                // tile_dim0 = 32
  g1[4] = 256;                                                        // tile_dim1 = 256
  g1[5] = ldw;                                                        // dim0 stride
  g1[6] = 0;
  g1[7] = 0;
  const unsigned long long wbase =
      (unsigned long long)(size_t)W + (unsigned long long)nblk * (unsigned)ldw * 2ull;
  const unsigned lds0 = (unsigned)(size_t)&wt[0][0];
  const unsigned lds1 = (unsigned)(size_t)&wt[1][0];
  const v4i z4 = {0, 0, 0, 0};
  const v8i z8 = {0, 0, 0, 0, 0, 0, 0, 0};
  auto tdm_issue = [&](int k0, int bufsel) {
    unsigned long long ga = wbase + (unsigned long long)(unsigned)k0 * 2ull;
    v4u g0;
    g0[0] = 1u;                                  // count = 1 (valid descriptor)
    g0[1] = bufsel ? lds1 : lds0;                // lds_addr
    g0[2] = (unsigned)ga;                        // global_addr lo
    g0[3] = (unsigned)(ga >> 32) | 0x80000000u;  // global_addr hi | type=2
    __builtin_amdgcn_tensor_load_to_lds(g0, g1, z4, z4, z8, 0);
  };
  if (w == 0) {
    tdm_issue(0, 0);
    __builtin_amdgcn_s_wait_tensorcnt(0);
  }
  __syncthreads();
  int buf = 0;
#endif

  for (int k0 = 0; k0 < K; k0 += 32) {
#if HAVE_TDM
    const char* wl = &wt[buf][0];
    if (w == 0 && k0 + 32 < K) tdm_issue(k0 + 32, buf ^ 1);
#else
    // Fallback: cooperative copy, one row per thread (same padded layout).
    __syncthreads();
    {
      const __bf16* src = W + (size_t)(nblk + threadIdx.x) * ldw + k0;
      uint4* dst = (uint4*)(&wt[0][0] + threadIdx.x * 80);
      const uint4* s4 = (const uint4*)src;
      dst[0] = s4[0]; dst[1] = s4[1]; dst[2] = s4[2]; dst[3] = s4[3];
    }
    __syncthreads();
    const char* wl = &wt[0][0];
#endif

    v8bf alo = *(const v8bf*)(Arow + k0 + hi * 8);
    v8bf ahi = *(const v8bf*)(Arow + k0 + 16 + hi * 8);
    v16bf af = cat8(alo, ahi);
    if (k0 + 32 < K) __builtin_prefetch(Arow + k0 + 64, 0, 1);

#pragma unroll
    for (int jj = 0; jj < 4; ++jj) {
      const char* wrow = wl + (nw + jj * 16 + li) * 80 + hi * 16;
      v8bf wlo = *(const v8bf*)(wrow);
      v8bf whi = *(const v8bf*)(wrow + 32);
      v16bf bfm = cat8(wlo, whi);
      acc[jj] = __builtin_amdgcn_wmma_f32_16x16x32_bf16(
          false, af, false, bfm, (short)0, acc[jj], false, false);
    }

#if HAVE_TDM
    if (k0 + 32 < K) {
      if (w == 0) __builtin_amdgcn_s_wait_tensorcnt(0);
      __syncthreads();
      buf ^= 1;
    }
#endif
  }

#pragma unroll
  for (int jj = 0; jj < 4; ++jj) {
    const int col = nblk + nw + jj * 16 + li;
    const float bv = bias ? bias[col] : 0.0f;
#pragma unroll
    for (int r = 0; r < 8; ++r) {
      const int row = m0 + hi * 8 + r;
      float v = acc[jj][r] + bv;
      const size_t o = (size_t)row * ldc + col;
      if (EPI == 0) {
        Cf[o] = v;
      } else if (EPI == 1) {
        Cf[o] = v + resid[(size_t)row * ldr + col];
      } else {  // GEGLU: v is gate(+bias); pre holds 'a'
        float a = pre[(size_t)row * ldp + col];
        float g = 0.5f * v * (1.0f + erff(v * 0.70710678118654752f));
        Cb[o] = (__bf16)(a * g);
      }
    }
  }
}

// ---------------------------------------------------------------------------
// Channel LayerNorm per token (512 contiguous), optional time-emb modulation.
// One wave per token; 8 tokens per 256-thread block. Emits bf16.
// ---------------------------------------------------------------------------
__global__ __launch_bounds__(256) void ln_kernel(
    const float* __restrict__ X, const float* __restrict__ g,
    const float* __restrict__ te,  // (B, 1024) scale|shift, or nullptr
    __bf16* __restrict__ out, int ldo, int col_off) {
  const int w = threadIdx.x >> 5, lane = threadIdx.x & 31;
  const int t = blockIdx.x * 8 + w;
  const float* xp = X + (size_t)t * 512 + lane * 16;
  float vals[16];
  float s = 0.f, s2 = 0.f;
#pragma unroll
  for (int i = 0; i < 16; ++i) {
    float v = xp[i];
    vals[i] = v; s += v; s2 += v * v;
  }
  s = wave_sum(s); s2 = wave_sum(s2);
  const float mu  = s * (1.0f / 512.0f);
  const float var = s2 * (1.0f / 512.0f) - mu * mu;
  const float rs  = rsqrtf(var + 1e-5f);
  __bf16* op = out + (size_t)t * ldo + col_off + lane * 16;
  const float* tep = te ? (te + (size_t)(t >> 6) * 1024 + lane * 16) : nullptr;
#pragma unroll
  for (int i = 0; i < 16; ++i) {
    const int c = lane * 16 + i;
    float y = (vals[i] - mu) * rs * g[c];
    if (te) y = y * (tep[i] + 1.0f) + tep[512 + i];
    op[i] = (__bf16)y;
  }
}

// ---------------------------------------------------------------------------
// Softmax attention per (batch, head): 64 tokens, DH=64.  WMMA for both the
// score GEMM (q^T k) and the output GEMM (attn * v^T).  128 threads = 4 waves.
// Y is the qkv buffer, token-major ld=1536.  Writes bf16 out (ld=512).
// ---------------------------------------------------------------------------
__global__ __launch_bounds__(128) void sa_attn_kernel(
    const float* __restrict__ Y, __bf16* __restrict__ out) {
  __shared__ float  sim[64 * 65];
  __shared__ __bf16 attn[64 * 80];
  const int b = blockIdx.x >> 3, h = blockIdx.x & 7;
  const float* Yb = Y + (size_t)b * 64 * 1536;
  const int tid = threadIdx.x, w = tid >> 5, lane = tid & 31;
  const int li = lane & 15, hi = lane >> 4;
  const int i0 = w * 16;

  // ---- scores: sim[i][j] = sum_d q[d,i] k[d,j] -------------------------
  v8f acc[4] = {v8f{}, v8f{}, v8f{}, v8f{}};
  const float* qp = Yb + (size_t)(i0 + li) * 1536 + h * 64;
#pragma unroll
  for (int k0 = 0; k0 < 64; k0 += 32) {
    v16bf af;
#pragma unroll
    for (int i = 0; i < 8; ++i) {
      af[i]     = (__bf16)qp[k0 + hi * 16 + i];
      af[8 + i] = (__bf16)qp[k0 + hi * 16 + 8 + i];
    }
#pragma unroll
    for (int jj = 0; jj < 4; ++jj) {
      const float* kp = Yb + (size_t)(jj * 16 + li) * 1536 + 512 + h * 64 + k0 + hi * 16;
      v16bf bfm;
#pragma unroll
      for (int i = 0; i < 16; ++i) bfm[i] = (__bf16)kp[i];
      acc[jj] = __builtin_amdgcn_wmma_f32_16x16x32_bf16(
          false, af, false, bfm, (short)0, acc[jj], false, false);
    }
  }
#pragma unroll
  for (int jj = 0; jj < 4; ++jj)
#pragma unroll
    for (int r = 0; r < 8; ++r)
      sim[(i0 + hi * 8 + r) * 65 + jj * 16 + li] = acc[jj][r];
  __syncthreads();

  // ---- softmax rows (SCALE = 1/8) --------------------------------------
  if (tid < 64) {
    float m = -1e30f;
    for (int j = 0; j < 64; ++j) m = fmaxf(m, sim[tid * 65 + j] * 0.125f);
    float s = 0.f;
    for (int j = 0; j < 64; ++j) s += expf(sim[tid * 65 + j] * 0.125f - m);
    const float inv = 1.0f / s;
    for (int j = 0; j < 64; ++j)
      attn[tid * 80 + j] = (__bf16)(expf(sim[tid * 65 + j] * 0.125f - m) * inv);
  }
  __syncthreads();

  // ---- out: o[i][d] = sum_j attn[i,j] v[d,j] ---------------------------
  v8f acc2[4] = {v8f{}, v8f{}, v8f{}, v8f{}};
  const __bf16* ap = attn + (size_t)(i0 + li) * 80;
#pragma unroll
  for (int k0 = 0; k0 < 64; k0 += 32) {
    v8bf alo = *(const v8bf*)(ap + k0 + hi * 8);
    v8bf ahi = *(const v8bf*)(ap + k0 + 16 + hi * 8);
    v16bf af = cat8(alo, ahi);
#pragma unroll
    for (int jj = 0; jj < 4; ++jj) {
      v16bf bfm;
#pragma unroll
      for (int i = 0; i < 16; ++i)
        bfm[i] = (__bf16)Yb[(size_t)(k0 + hi * 8 + (i & 7) + (i >> 3) * 16) * 1536 + 1024 + h * 64 + jj * 16 + li];
      acc2[jj] = __builtin_amdgcn_wmma_f32_16x16x32_bf16(
          false, af, false, bfm, (short)0, acc2[jj], false, false);
    }
  }
#pragma unroll
  for (int jj = 0; jj < 4; ++jj)
#pragma unroll
    for (int r = 0; r < 8; ++r)
      out[(size_t)(b * 64 + i0 + hi * 8 + r) * 512 + h * 64 + jj * 16 + li] =
          (__bf16)acc2[jj][r];
}

// ---------------------------------------------------------------------------
// Linear attention per (batch, head): q softmax over d, k softmax over n,
// ctx = k v^T / 64, out = ctx^T q.  64 threads/block; tiny, VALU is fine.
// ---------------------------------------------------------------------------
__global__ __launch_bounds__(64) void la_math_kernel(
    const float* __restrict__ Y, __bf16* __restrict__ out) {
  __shared__ float lq[64 * 65], lk[64 * 65], lv[64 * 65];
  const int b = blockIdx.x >> 3, h = blockIdx.x & 7;
  const float* Yb = Y + (size_t)b * 64 * 1536;
  const int t = threadIdx.x;

  {  // q: token t, softmax over d, times SCALE=1/8
    const float* p = Yb + (size_t)t * 1536 + h * 64;
    float m = -1e30f;
    for (int d = 0; d < 64; ++d) m = fmaxf(m, p[d]);
    float s = 0.f;
    for (int d = 0; d < 64; ++d) s += expf(p[d] - m);
    const float inv = 0.125f / s;
    for (int d = 0; d < 64; ++d) lq[d * 65 + t] = expf(p[d] - m) * inv;
  }
  {  // k: channel d=t, softmax over tokens n
    const float* p = Yb + 512 + h * 64 + t;
    float m = -1e30f;
    for (int n = 0; n < 64; ++n) m = fmaxf(m, p[(size_t)n * 1536]);
    float s = 0.f;
    for (int n = 0; n < 64; ++n) s += expf(p[(size_t)n * 1536] - m);
    const float inv = 1.0f / s;
    for (int n = 0; n < 64; ++n) lk[t * 65 + n] = expf(p[(size_t)n * 1536] - m) * inv;
  }
  {  // v / 64
    const float* p = Yb + (size_t)t * 1536 + 1024 + h * 64;
    for (int e = 0; e < 64; ++e) lv[e * 65 + t] = p[e] * (1.0f / 64.0f);
  }
  __syncthreads();

  float cr[64];
#pragma unroll
  for (int e = 0; e < 64; ++e) cr[e] = 0.f;
  for (int n = 0; n < 64; ++n) {
    const float kd = lk[t * 65 + n];
#pragma unroll
    for (int e = 0; e < 64; ++e) cr[e] += kd * lv[e * 65 + n];
  }
  __syncthreads();
#pragma unroll
  for (int e = 0; e < 64; ++e) lk[t * 65 + e] = cr[e];  // lk := ctx[d=t][e]
  __syncthreads();

  float o[64];
#pragma unroll
  for (int e = 0; e < 64; ++e) o[e] = 0.f;
  for (int d = 0; d < 64; ++d) {
    const float qd = lq[d * 65 + t];
#pragma unroll
    for (int e = 0; e < 64; ++e) o[e] += lk[d * 65 + e] * qd;
  }
  __bf16* op = out + (size_t)(b * 64 + t) * 512 + h * 64;
#pragma unroll
  for (int e = 0; e < 64; ++e) op[e] = (__bf16)o[e];
}

// ---------------------------------------------------------------------------
// Small elementwise kernels
// ---------------------------------------------------------------------------
__global__ void cvt_f32_bf16(const float* __restrict__ s, __bf16* __restrict__ d, int n) {
  int i = blockIdx.x * 256 + threadIdx.x;
  if (i < n) d[i] = (__bf16)s[i];
}
__global__ void silu_cvt_kernel(const float* __restrict__ s, __bf16* __restrict__ d) {
  int i = blockIdx.x * 256 + threadIdx.x;
  float v = s[i];
  d[i] = (__bf16)(v / (1.0f + expf(-v)));
}
__global__ void trans_in_kernel(const float* __restrict__ x, float* __restrict__ XF) {
  int idx = blockIdx.x * 256 + threadIdx.x;      // b*2^15 + c*64 + hw
  int b = idx >> 15, c = (idx >> 6) & 511, hw = idx & 63;
  XF[(size_t)(b * 64 + hw) * 512 + c] = x[idx];
}
__global__ void trans_out_kernel(const float* __restrict__ XF, float* __restrict__ o) {
  int idx = blockIdx.x * 256 + threadIdx.x;
  int b = idx >> 15, c = (idx >> 6) & 511, hw = idx & 63;
  o[idx] = XF[(size_t)(b * 64 + hw) * 512 + c];
}
__global__ void bc_vc_kernel(const float* __restrict__ VC, __bf16* __restrict__ XN) {
  int idx = blockIdx.x * 256 + threadIdx.x;      // token*512 + c
  int tok = idx >> 9, c = idx & 511;
  XN[idx] = (__bf16)VC[(size_t)(tok >> 6) * 512 + c];
}

// ---------------------------------------------------------------------------
extern "C" void kernel_launch(void* const* d_in, const int* in_sizes, int n_in,
                              void* d_out, int out_size, void* d_ws, size_t ws_size,
                              hipStream_t stream) {
  (void)in_sizes; (void)n_in; (void)out_size; (void)ws_size;
  const float* x         = (const float*)d_in[0];
  const float* context   = (const float*)d_in[1];
  const float* time_emb  = (const float*)d_in[2];
  const float* g1        = (const float*)d_in[3];
  const float* g4        = (const float*)d_in[4];
  const float* g6        = (const float*)d_in[6];   // g5 is dead (CA shortcut)
  const float* sa_w_qkv  = (const float*)d_in[7];
  const float* sa_w_out  = (const float*)d_in[8];
  const float* sa_b_out  = (const float*)d_in[9];
  const float* sa_w_mlp  = (const float*)d_in[10];
  const float* sa_b_mlp  = (const float*)d_in[11];
  const float* laf_w_qkv = (const float*)d_in[12];
  const float* laf_w_out = (const float*)d_in[13];
  const float* laf_b_out = (const float*)d_in[14];
  const float* laf_g     = (const float*)d_in[15];
  const float* lat_w_qkv = (const float*)d_in[16];
  const float* lat_w_out = (const float*)d_in[17];
  const float* lat_b_out = (const float*)d_in[18];
  const float* lat_g     = (const float*)d_in[19];
  const float* la_w_comb = (const float*)d_in[20];
  const float* la_b_comb = (const float*)d_in[21];
  const float* ca_w_v    = (const float*)d_in[26];  // ca_w_q/ca_w_k dead
  const float* ca_b_v    = (const float*)d_in[27];
  const float* ca_w_out  = (const float*)d_in[28];
  const float* ca_b_out  = (const float*)d_in[29];
  const float* ff_w_proj = (const float*)d_in[30];
  const float* ff_b_proj = (const float*)d_in[31];
  const float* ff_w_out  = (const float*)d_in[32];
  const float* ff_b_out  = (const float*)d_in[33];

  char* ws = (char*)d_ws;
  float*  XF   = (float*)(ws + 0);                 // 16384x512 f32
  __bf16* XN1  = (__bf16*)(ws + 33554432);         // 16384x512 bf16
  __bf16* XN2  = (__bf16*)(ws + 50331648);         // 16384x2048 bf16
  __bf16* XN3  = (__bf16*)(ws + 117440512);        // 16384x512 bf16
  float*  Y    = (float*)(ws + 134217728);         // 16384x2048 f32
  float*  T    = (float*)(ws + 268435456);         // 16384x512 f32
  float*  TE   = (float*)(ws + 301989888);         // 256x1024 f32
  __bf16* SIL  = (__bf16*)(ws + 303038464);        // 256x512 bf16
  __bf16* CTXB = (__bf16*)(ws + 303300608);        // 256x512 bf16
  float*  VC   = (float*)(ws + 303562752);         // 256x512 f32
  __bf16* WB   = (__bf16*)(ws + 304087040);        // packed bf16 weights

  __bf16* w_sa_qkv  = WB + 0;
  __bf16* w_sa_out  = WB + 786432;
  __bf16* w_sa_mlp  = WB + 1048576;
  __bf16* w_laf_qkv = WB + 1572864;
  __bf16* w_laf_out = WB + 2359296;
  __bf16* w_lat_qkv = WB + 2621440;
  __bf16* w_lat_out = WB + 3407872;
  __bf16* w_comb    = WB + 3670016;
  __bf16* w_ca_v    = WB + 4194304;
  __bf16* w_ca_out  = WB + 4456448;
  __bf16* w_ff_proj = WB + 4718592;
  __bf16* w_ff_out  = WB + 6815744;

  auto cvt = [&](const float* s, __bf16* d, int n) {
    cvt_f32_bf16<<<(n + 255) / 256, 256, 0, stream>>>(s, d, n);
  };
  cvt(sa_w_qkv,  w_sa_qkv,  786432);
  cvt(sa_w_out,  w_sa_out,  262144);
  cvt(sa_w_mlp,  w_sa_mlp,  524288);
  cvt(laf_w_qkv, w_laf_qkv, 786432);
  cvt(laf_w_out, w_laf_out, 262144);
  cvt(lat_w_qkv, w_lat_qkv, 786432);
  cvt(lat_w_out, w_lat_out, 262144);
  cvt(la_w_comb, w_comb,    524288);
  cvt(ca_w_v,    w_ca_v,    262144);
  cvt(ca_w_out,  w_ca_out,  262144);
  cvt(ff_w_proj, w_ff_proj, 2097152);
  cvt(ff_w_out,  w_ff_out,  1048576);

  trans_in_kernel<<<32768, 256, 0, stream>>>(x, XF);
  silu_cvt_kernel<<<512, 256, 0, stream>>>(time_emb, SIL);
  cvt(context, CTXB, 131072);

  const dim3 gFull512(2, 512), gFull1536(6, 512), gFull2048(8, 512);

  // time-embedding MLP: (256x512)x(512x1024)
  gemm_bf16_kernel<0><<<dim3(4, 8), 256, 0, stream>>>(
      SIL, 512, w_sa_mlp, 512, 1024, sa_b_mlp, nullptr, 0, nullptr, 0, TE, nullptr, 1024, 512);

  // --- self attention ---
  ln_kernel<<<2048, 256, 0, stream>>>(XF, g1, TE, XN1, 512, 0);
  gemm_bf16_kernel<0><<<gFull1536, 256, 0, stream>>>(
      XN1, 512, w_sa_qkv, 512, 1536, nullptr, nullptr, 0, nullptr, 0, Y, nullptr, 1536, 512);
  sa_attn_kernel<<<2048, 128, 0, stream>>>(Y, XN1);
  gemm_bf16_kernel<1><<<gFull512, 256, 0, stream>>>(
      XN1, 512, w_sa_out, 512, 512, sa_b_out, XF, 512, nullptr, 0, XF, nullptr, 512, 512);

  // --- linear attention (f branch then t branch) ---
  ln_kernel<<<2048, 256, 0, stream>>>(XF, g4, nullptr, XN1, 512, 0);

  gemm_bf16_kernel<0><<<gFull1536, 256, 0, stream>>>(
      XN1, 512, w_laf_qkv, 512, 1536, nullptr, nullptr, 0, nullptr, 0, Y, nullptr, 1536, 512);
  la_math_kernel<<<2048, 64, 0, stream>>>(Y, XN3);
  gemm_bf16_kernel<0><<<gFull512, 256, 0, stream>>>(
      XN3, 512, w_laf_out, 512, 512, laf_b_out, nullptr, 0, nullptr, 0, T, nullptr, 512, 512);
  ln_kernel<<<2048, 256, 0, stream>>>(T, laf_g, nullptr, XN2, 1024, 0);

  gemm_bf16_kernel<0><<<gFull1536, 256, 0, stream>>>(
      XN1, 512, w_lat_qkv, 512, 1536, nullptr, nullptr, 0, nullptr, 0, Y, nullptr, 1536, 512);
  la_math_kernel<<<2048, 64, 0, stream>>>(Y, XN3);
  gemm_bf16_kernel<0><<<gFull512, 256, 0, stream>>>(
      XN3, 512, w_lat_out, 512, 512, lat_b_out, nullptr, 0, nullptr, 0, T, nullptr, 512, 512);
  ln_kernel<<<2048, 256, 0, stream>>>(T, lat_g, nullptr, XN2, 1024, 512);

  gemm_bf16_kernel<1><<<gFull512, 256, 0, stream>>>(
      XN2, 1024, w_comb, 1024, 512, la_b_comb, XF, 512, nullptr, 0, XF, nullptr, 512, 1024);

  // --- cross attention (keys identical => out == vc exactly) ---
  gemm_bf16_kernel<0><<<dim3(2, 8), 256, 0, stream>>>(
      CTXB, 512, w_ca_v, 512, 512, ca_b_v, nullptr, 0, nullptr, 0, VC, nullptr, 512, 512);
  bc_vc_kernel<<<32768, 256, 0, stream>>>(VC, XN1);
  gemm_bf16_kernel<1><<<gFull512, 256, 0, stream>>>(
      XN1, 512, w_ca_out, 512, 512, ca_b_out, XF, 512, nullptr, 0, XF, nullptr, 512, 512);

  // --- GEGLU feed-forward ---
  ln_kernel<<<2048, 256, 0, stream>>>(XF, g6, nullptr, XN1, 512, 0);
  gemm_bf16_kernel<0><<<gFull2048, 256, 0, stream>>>(        // 'a' half
      XN1, 512, w_ff_proj, 512, 2048, ff_b_proj, nullptr, 0, nullptr, 0, Y, nullptr, 2048, 512);
  gemm_bf16_kernel<2><<<gFull2048, 256, 0, stream>>>(        // gate half + fuse
      XN1, 512, w_ff_proj + (size_t)2048 * 512, 512, 2048, ff_b_proj + 2048,
      nullptr, 0, Y, 2048, nullptr, XN2, 2048, 512);
  gemm_bf16_kernel<1><<<gFull512, 256, 0, stream>>>(
      XN2, 2048, w_ff_out, 2048, 512, ff_b_out, XF, 512, nullptr, 0, XF, nullptr, 512, 2048);

  trans_out_kernel<<<32768, 256, 0, stream>>>(XF, (float*)d_out);
}
